// Attention_21835613733572
// MI455X (gfx1250) — matrix operand order repaired and checked by reading, not compiled, not run on hardware
//
#include <hip/hip_runtime.h>

#define BB 16
#define CC 512
#define LL 1024

typedef __attribute__((ext_vector_type(16))) __bf16 v16bf;
typedef __attribute__((ext_vector_type(8)))  float  v8f;

union Frag {
    uint4 u[2];
    v16bf v;
};

__device__ __forceinline__ unsigned short f2bf(float x) {
    unsigned u = __float_as_uint(x);
    u += 0x7FFFu + ((u >> 16) & 1u);      // round-to-nearest-even
    return (unsigned short)(u >> 16);
}
__device__ __forceinline__ float bf2f(unsigned short h) {
    return __uint_as_float(((unsigned)h) << 16);
}

// ---------------- async global->LDS 16-byte copy (gfx1250 ASYNCcnt path) ----------------
#if __has_builtin(__builtin_amdgcn_global_load_async_to_lds_b128)
#define HAVE_ASYNC_CP 1
#endif

typedef int v4i __attribute__((vector_size(16)));
typedef __attribute__((address_space(1))) v4i* gv4i_p;   // global (device) addrspace
typedef __attribute__((address_space(3))) v4i* lv4i_p;   // LDS (shared) addrspace

__device__ __forceinline__ void async_cp16(const void* g, void* l) {
#ifdef HAVE_ASYNC_CP
    __builtin_amdgcn_global_load_async_to_lds_b128(
        (gv4i_p)(uintptr_t)g, (lv4i_p)(unsigned)(uintptr_t)l, 0, 0);
#else
    *(uint4*)l = *(const uint4*)g;
#endif
}

__device__ __forceinline__ void wait_async0() {
#ifdef HAVE_ASYNC_CP
#if __has_builtin(__builtin_amdgcn_s_wait_asynccnt)
    __builtin_amdgcn_s_wait_asynccnt(0);
#else
    asm volatile("s_wait_asynccnt 0" ::: "memory");
#endif
#endif
}

// ---------------------------------------------------------------- cvt+transpose: q,k -> bf16 (L x C)
// qT[b][l][c] = bf16(q[b][c][l]);  kT likewise.  grid: (L/32, C/32, B), block 256
__global__ __launch_bounds__(256) void cvtT_kernel(const float* __restrict__ q,
                                                   const float* __restrict__ k,
                                                   unsigned short* __restrict__ qT,
                                                   unsigned short* __restrict__ kT) {
    __shared__ unsigned short tq[32][33];
    __shared__ unsigned short tk[32][33];
    const int tid = threadIdx.x;
    const int tx = tid & 31, ty = tid >> 5;   // ty 0..7
    const int b  = blockIdx.z;
    const int c0 = blockIdx.y * 32;
    const int l0 = blockIdx.x * 32;
    #pragma unroll
    for (int cy = 0; cy < 4; ++cy) {
        int cl = cy * 8 + ty;
        size_t base = (size_t)(b * CC + c0 + cl) * LL + l0 + tx;
        tq[cl][tx] = f2bf(q[base]);
        tk[cl][tx] = f2bf(k[base]);
    }
    __syncthreads();
    #pragma unroll
    for (int cy = 0; cy < 4; ++cy) {
        int rl = cy * 8 + ty;
        size_t base = (size_t)(b * LL + l0 + rl) * CC + c0 + tx;
        qT[base] = tq[tx][rl];
        kT[base] = tk[tx][rl];
    }
}

// ---------------------------------------------------------------- cvt: v -> bf16 (layout preserved)
__global__ void cvtV_kernel(const float* __restrict__ v, unsigned short* __restrict__ vb) {
    size_t i4 = ((size_t)blockIdx.x * blockDim.x + threadIdx.x) * 4;
    float4 f = *(const float4*)(v + i4);
    uint2 o;
    o.x = (unsigned)f2bf(f.x) | ((unsigned)f2bf(f.y) << 16);
    o.y = (unsigned)f2bf(f.z) | ((unsigned)f2bf(f.w) << 16);
    *(uint2*)(vb + i4) = o;
}

// ---------------------------------------------------------------- score: sT[b][j][i] = (1/32) sum_c kT[b][i][c]*qT[b][j][c]
// grid: (L/128 j, L/128 i, B), block 256 (8 waves), double-buffered async staging
__global__ __launch_bounds__(256) void score_kernel(const unsigned short* __restrict__ kT,
                                                    const unsigned short* __restrict__ qT,
                                                    unsigned short* __restrict__ sT) {
    __shared__ unsigned short ks[2][128][40];   // [buf][i][c] padded rows (80B)
    __shared__ unsigned short qs[2][128][40];   // [buf][j][c]

    const int tid  = threadIdx.x;
    const int lane = tid & 31;
    const int wid  = tid >> 5;
    const int b    = blockIdx.z;
    const int i0   = blockIdx.y * 128;
    const int j0   = blockIdx.x * 128;
    const int half = lane >> 4;
    const int lrow = lane & 15;

    const unsigned short* kbase = kT + (size_t)(b * LL + i0) * CC;
    const unsigned short* qbase = qT + (size_t)(b * LL + j0) * CC;

    v8f acc[8];
    #pragma unroll
    for (int t = 0; t < 8; ++t)
        #pragma unroll
        for (int r = 0; r < 8; ++r) acc[t][r] = 0.f;

    auto stage = [&](int buf, int c0) {
        #pragma unroll
        for (int it = 0; it < 2; ++it) {
            int idx = tid + it * 256;          // 0..511
            int row = idx >> 2;                // 0..127
            int seg = (idx & 3) * 8;           // column segment (8 bf16 = 16B)
            async_cp16(kbase + (size_t)row * CC + c0 + seg, &ks[buf][row][seg]);
            async_cp16(qbase + (size_t)row * CC + c0 + seg, &qs[buf][row][seg]);
        }
    };

    stage(0, 0);
    int cur = 0;
    for (int c0 = 0; c0 < CC; c0 += 32) {
        wait_async0();
        __syncthreads();
        if (c0 + 32 < CC) stage(cur ^ 1, c0 + 32);

        Frag fb;
        {
            int row = wid * 16 + lrow;
            int cb  = half * 8;
            fb.u[0] = *(const uint4*)&qs[cur][row][cb];
            fb.u[1] = *(const uint4*)&qs[cur][row][cb + 16];
        }
        #pragma unroll
        for (int t = 0; t < 8; ++t) {
            Frag fa;
            int row = t * 16 + lrow;
            int cb  = half * 8;
            fa.u[0] = *(const uint4*)&ks[cur][row][cb];
            fa.u[1] = *(const uint4*)&ks[cur][row][cb + 16];
            acc[t] = __builtin_amdgcn_wmma_f32_16x16x32_bf16(
                false, fa.v, false, fb.v, (short)0, acc[t], false, false);
        }
        cur ^= 1;
    }

    // packed b128 epilogue: lane's 8 accum elements are contiguous in i
    const float scale = 0.03125f;  // 1/sqrt(1024)
    const int gj = j0 + wid * 16 + lrow;
    #pragma unroll
    for (int t = 0; t < 8; ++t) {
        int gi = i0 + t * 16 + half * 8;
        uint4 o;
        o.x = (unsigned)f2bf(acc[t][0] * scale) | ((unsigned)f2bf(acc[t][1] * scale) << 16);
        o.y = (unsigned)f2bf(acc[t][2] * scale) | ((unsigned)f2bf(acc[t][3] * scale) << 16);
        o.z = (unsigned)f2bf(acc[t][4] * scale) | ((unsigned)f2bf(acc[t][5] * scale) << 16);
        o.w = (unsigned)f2bf(acc[t][6] * scale) | ((unsigned)f2bf(acc[t][7] * scale) << 16);
        *(uint4*)(sT + ((size_t)(b * LL + gj) * LL + gi)) = o;
    }
}

// ---------------------------------------------------------------- softmax over batch axis (16), in place, 2 positions/thread
__global__ void softmax_kernel(unsigned short* __restrict__ sb) {
    size_t p2 = ((size_t)blockIdx.x * blockDim.x + threadIdx.x) * 2;
    const size_t stride = (size_t)LL * LL;
    float a[BB], c[BB];
    float m0 = -1e30f, m1 = -1e30f;
    #pragma unroll
    for (int b = 0; b < BB; ++b) {
        unsigned u = *(const unsigned*)(sb + (size_t)b * stride + p2);
        a[b] = bf2f((unsigned short)(u & 0xFFFFu));
        c[b] = bf2f((unsigned short)(u >> 16));
        m0 = fmaxf(m0, a[b]);
        m1 = fmaxf(m1, c[b]);
    }
    float s0 = 0.f, s1 = 0.f;
    #pragma unroll
    for (int b = 0; b < BB; ++b) {
        a[b] = __expf(a[b] - m0); s0 += a[b];
        c[b] = __expf(c[b] - m1); s1 += c[b];
    }
    float i0 = 1.f / s0, i1 = 1.f / s1;
    #pragma unroll
    for (int b = 0; b < BB; ++b) {
        unsigned u = (unsigned)f2bf(a[b] * i0) | ((unsigned)f2bf(c[b] * i1) << 16);
        *(unsigned*)(sb + (size_t)b * stride + p2) = u;
    }
}

// ---------------------------------------------------------------- out[b][c][j] = sum_i vb[b][c][i] * wT[b][j][i]
// grid: (L/128 j, C/128 c, B), block 256, double-buffered async staging
__global__ __launch_bounds__(256) void out_kernel(const unsigned short* __restrict__ vb,
                                                  const unsigned short* __restrict__ wT,
                                                  float* __restrict__ out) {
    __shared__ unsigned short as_[2][128][40];  // [buf][c][i-chunk]
    __shared__ unsigned short bs[2][128][40];   // [buf][j][i-chunk]

    const int tid  = threadIdx.x;
    const int lane = tid & 31;
    const int wid  = tid >> 5;
    const int b    = blockIdx.z;
    const int c0   = blockIdx.y * 128;
    const int j0   = blockIdx.x * 128;
    const int half = lane >> 4;
    const int lrow = lane & 15;

    const unsigned short* abase = vb + (size_t)(b * CC + c0) * LL;
    const unsigned short* bbase = wT + (size_t)(b * LL + j0) * LL;

    v8f acc[8];
    #pragma unroll
    for (int t = 0; t < 8; ++t)
        #pragma unroll
        for (int r = 0; r < 8; ++r) acc[t][r] = 0.f;

    auto stage = [&](int buf, int i0c) {
        #pragma unroll
        for (int it = 0; it < 2; ++it) {
            int idx = tid + it * 256;
            int row = idx >> 2;
            int seg = (idx & 3) * 8;
            async_cp16(abase + (size_t)row * LL + i0c + seg, &as_[buf][row][seg]);
            async_cp16(bbase + (size_t)row * LL + i0c + seg, &bs[buf][row][seg]);
        }
    };

    stage(0, 0);
    int cur = 0;
    for (int i0c = 0; i0c < LL; i0c += 32) {
        wait_async0();
        __syncthreads();
        if (i0c + 32 < LL) stage(cur ^ 1, i0c + 32);

        Frag fb;
        {
            int row = wid * 16 + lrow;
            int cb  = half * 8;
            fb.u[0] = *(const uint4*)&bs[cur][row][cb];
            fb.u[1] = *(const uint4*)&bs[cur][row][cb + 16];
        }
        #pragma unroll
        for (int t = 0; t < 8; ++t) {
            Frag fa;
            int row = t * 16 + lrow;
            int cb  = half * 8;
            fa.u[0] = *(const uint4*)&as_[cur][row][cb];
            fa.u[1] = *(const uint4*)&as_[cur][row][cb + 16];
            acc[t] = __builtin_amdgcn_wmma_f32_16x16x32_bf16(
                false, fa.v, false, fb.v, (short)0, acc[t], false, false);
        }
        cur ^= 1;
    }

    #pragma unroll
    for (int t = 0; t < 8; ++t) {
        #pragma unroll
        for (int r = 0; r < 8; ++r) {
            int gc = c0 + t * 16 + r + half * 8;
            int gj = j0 + wid * 16 + lrow;
            out[((size_t)b * CC + gc) * LL + gj] = acc[t][r];
        }
    }
}

// ----------------------------------------------------------------
extern "C" void kernel_launch(void* const* d_in, const int* in_sizes, int n_in,
                              void* d_out, int out_size, void* d_ws, size_t ws_size,
                              hipStream_t stream) {
    const float* q = (const float*)d_in[0];
    const float* k = (const float*)d_in[1];
    const float* v = (const float*)d_in[2];
    float* out = (float*)d_out;

    char* ws = (char*)d_ws;
    unsigned short* qT = (unsigned short*)(ws);                                  // 16 MB  (L x C per b)
    unsigned short* kT = (unsigned short*)(ws + (size_t)16 * 1024 * 1024);       // 16 MB  (L x C per b)
    unsigned short* vb = (unsigned short*)(ws + (size_t)32 * 1024 * 1024);       // 16 MB  (C x L per b)
    unsigned short* sT = (unsigned short*)(ws + (size_t)48 * 1024 * 1024);       // 32 MB  scores/weights [b][j][i]

    cvtT_kernel<<<dim3(LL / 32, CC / 32, BB), 256, 0, stream>>>(q, k, qT, kT);
    cvtV_kernel<<<(unsigned)((size_t)BB * CC * LL / 4 / 256), 256, 0, stream>>>(v, vb);
    score_kernel<<<dim3(LL / 128, LL / 128, BB), 256, 0, stream>>>(kT, qT, sT);
    softmax_kernel<<<(LL * LL / 2) / 256, 256, 0, stream>>>(sT);
    out_kernel<<<dim3(LL / 128, CC / 128, BB), 256, 0, stream>>>(vb, sT, out);
}
